// PointNetPlusClassify_7576322310712
// MI455X (gfx1250) — matrix operand order repaired and checked by reading, not compiled, not run on hardware
//
#include <hip/hip_runtime.h>

typedef _Float16 f16;
typedef __attribute__((ext_vector_type(16))) _Float16 v16h;
typedef __attribute__((ext_vector_type(8)))  _Float16 v8h;
typedef __attribute__((ext_vector_type(8)))  float    v8f;

// ---------------------------------------------------------------------------
// (B,3,N) -> (B,N,3)
__global__ void k_transpose(const float* __restrict__ pc, float* __restrict__ xyz,
                            int B, int N) {
  int i = blockIdx.x * blockDim.x + threadIdx.x;
  int total = B * 3 * N;
  if (i >= total) return;
  int b = i / (3 * N);
  int rem = i - b * 3 * N;
  int c = rem / N;
  int n = rem - c * N;
  xyz[((size_t)b * N + n) * 3 + c] = pc[i];
}

// ---------------------------------------------------------------------------
// Farthest point sampling: one block (256 thr) per batch, dist array in LDS.
__global__ void k_fps(const float* __restrict__ xyz, int* __restrict__ fidx,
                      float* __restrict__ ctr, int N, int S) {
  __shared__ float dist[4096];
  __shared__ float rv[256];
  __shared__ int   ri[256];
  __shared__ int   s_far;
  const int b = blockIdx.x;
  const int t = threadIdx.x;
  const float* P = xyz + (size_t)b * N * 3;
  for (int i = t; i < N; i += 256) dist[i] = 1e10f;
  if (t == 0) s_far = 0;
  __syncthreads();
  for (int it = 0; it < S; ++it) {
    int far = s_far;
    float cx = P[far * 3 + 0], cy = P[far * 3 + 1], cz = P[far * 3 + 2];
    if (t == 0) {
      fidx[b * S + it] = far;
      ctr[((size_t)b * S + it) * 3 + 0] = cx;
      ctr[((size_t)b * S + it) * 3 + 1] = cy;
      ctr[((size_t)b * S + it) * 3 + 2] = cz;
    }
    float best = -1.f; int bidx = 0;
    for (int i = t; i < N; i += 256) {
      float dx = P[i * 3 + 0] - cx;
      float dy = P[i * 3 + 1] - cy;
      float dz = P[i * 3 + 2] - cz;
      float d = dx * dx + dy * dy + dz * dz;
      float nd = fminf(dist[i], d);
      dist[i] = nd;
      if (nd > best) { best = nd; bidx = i; }
    }
    rv[t] = best; ri[t] = bidx;
    __syncthreads();
    for (int o = 128; o > 0; o >>= 1) {
      if (t < o) {
        float ov = rv[t + o]; int oi = ri[t + o];
        if (ov > rv[t] || (ov == rv[t] && oi < ri[t])) { rv[t] = ov; ri[t] = oi; }
      }
      __syncthreads();
    }
    if (t == 0) s_far = ri[0];
    __syncthreads();
  }
}

// ---------------------------------------------------------------------------
// SA1 ball-query + group: wave per group; neighbor list in LDS, then each lane
// emits whole padded rows (32 ch: rel(3) + zeros) as two v16h stores.
// nf layout: (B*S*K rows, 32 ch) f16.
__global__ void k_group_sa1(const float* __restrict__ xyz, const float* __restrict__ ctr,
                            f16* __restrict__ nf, int B, int N, int S, int K, float r2) {
  __shared__ int slist[8][128];
  int wv = threadIdx.x >> 5;
  int lane = threadIdx.x & 31;
  int g = blockIdx.x * 8 + wv;
  if (g >= B * S) return;
  int b = g / S;
  const float* P = xyz + (size_t)b * N * 3;
  float cx = ctr[(size_t)g * 3 + 0];
  float cy = ctr[(size_t)g * 3 + 1];
  float cz = ctr[(size_t)g * 3 + 2];
  int cnt = 0, firsti = -1;
  for (int base = 0; base < N && cnt < K; base += 32) {
    int i = base + lane;
    bool ok = false;
    if (i < N) {
      float dx = P[i * 3 + 0] - cx;
      float dy = P[i * 3 + 1] - cy;
      float dz = P[i * 3 + 2] - cz;
      ok = (dx * dx + dy * dy + dz * dz) <= r2;
    }
    unsigned mask = __builtin_amdgcn_ballot_w32(ok);
    while (mask && cnt < K) {
      int j = __builtin_ctz(mask);
      mask &= mask - 1;
      int idx = base + j;
      if (firsti < 0) firsti = idx;
      slist[wv][cnt] = idx;   // wave-uniform value
      ++cnt;
    }
  }
  if (firsti < 0) firsti = N - 1;
  for (int kk = cnt; kk < K; ++kk) slist[wv][kk] = firsti;

  for (int kk = lane; kk < K; kk += 32) {
    int idx = slist[wv][kk];
    v16h lo = {};
    v16h hiv = {};
    lo[0] = (f16)(P[idx * 3 + 0] - cx);
    lo[1] = (f16)(P[idx * 3 + 1] - cy);
    lo[2] = (f16)(P[idx * 3 + 2] - cz);
    f16* row = nf + ((size_t)g * K + kk) * 32;
    *(v16h*)(row)      = lo;
    *(v16h*)(row + 16) = hiv;
  }
}

// ---------------------------------------------------------------------------
// SA2 ball-query + group with features. F layout: (B*N rows, 320 ch) f16.
// nf layout: (B*S*K rows, 352 ch) = rel(3) + feats(320) + zero pad(29).
__global__ void k_group_sa2(const float* __restrict__ xyz, const float* __restrict__ ctr,
                            const f16* __restrict__ F, f16* __restrict__ nf,
                            int B, int N, int S, int K, float r2) {
  __shared__ int slist[8][64];
  int wv = threadIdx.x >> 5;
  int lane = threadIdx.x & 31;
  int g = blockIdx.x * 8 + wv;
  if (g >= B * S) return;
  int b = g / S;
  const float* P = xyz + (size_t)b * N * 3;
  float cx = ctr[(size_t)g * 3 + 0];
  float cy = ctr[(size_t)g * 3 + 1];
  float cz = ctr[(size_t)g * 3 + 2];
  int cnt = 0, firsti = -1;
  for (int base = 0; base < N && cnt < K; base += 32) {
    int i = base + lane;
    bool ok = false;
    if (i < N) {
      float dx = P[i * 3 + 0] - cx;
      float dy = P[i * 3 + 1] - cy;
      float dz = P[i * 3 + 2] - cz;
      ok = (dx * dx + dy * dy + dz * dz) <= r2;
    }
    unsigned mask = __builtin_amdgcn_ballot_w32(ok);
    while (mask && cnt < K) {
      int j = __builtin_ctz(mask);
      mask &= mask - 1;
      int idx = base + j;
      if (firsti < 0) firsti = idx;
      slist[wv][cnt] = idx;
      ++cnt;
    }
  }
  if (firsti < 0) firsti = N - 1;
  for (int kk = cnt; kk < K; ++kk) slist[wv][kk] = firsti;

  for (int kk = 0; kk < K; ++kk) {
    int idx = slist[wv][kk];
    float rx = P[idx * 3 + 0] - cx;
    float ry = P[idx * 3 + 1] - cy;
    float rz = P[idx * 3 + 2] - cz;
    const f16* frow = F + ((size_t)b * N + idx) * 320;
    f16* row = nf + ((size_t)g * K + kk) * 352;
    for (int c = lane; c < 352; c += 32) {
      f16 v;
      if (c < 3)        v = (f16)(c == 0 ? rx : (c == 1 ? ry : rz));
      else if (c < 323) v = frow[c - 3];
      else              v = (f16)0.f;
      row[c] = v;
    }
  }
}

// ---------------------------------------------------------------------------
__global__ void k_zero(float* __restrict__ p, int n) {
  int i = blockIdx.x * blockDim.x + threadIdx.x;
  if (i < n) p[i] = 0.f;
}

// ---------------------------------------------------------------------------
// Weight convert+pad: W (Cout, Cin) f32 -> Wh (CoutP, CinP) f16, zero padded.
__global__ void k_cvt_w(const float* __restrict__ W, f16* __restrict__ Wh,
                        int Cout, int Cin, int CinP, int CoutP) {
  int i = blockIdx.x * blockDim.x + threadIdx.x;
  if (i >= CoutP * CinP) return;
  int o = i / CinP, c = i - o * CinP;
  Wh[i] = (o < Cout && c < Cin) ? (f16)W[(size_t)o * Cin + c] : (f16)0.f;
}

// ---------------------------------------------------------------------------
// WMMA GEMM with 4-way M register blocking (activation traffic /4).
//   Wh (CoutP, CinP) f16 zero-padded, CoutP multiple of 64,
//   X (N rows, CinP) f16,  Y (N rows, CoutP) f16,  N multiple of 16.
// Block = 8 waves; block tile = 64(M) x 128(N); wave = 4 accs of 16x16.
// Per K-step: 1x32B B load shared by 4 WMMAs + 4x2x16B A loads (L2-resident).
__global__ void k_gemm16(const f16* __restrict__ Wh, const f16* __restrict__ X,
                         f16* __restrict__ Y, float* __restrict__ stats,
                         int CinP, int CoutP, int Cout, int N) {
  __shared__ float s_sum[64];
  __shared__ float s_sq[64];
  if (threadIdx.x < 64) { s_sum[threadIdx.x] = 0.f; s_sq[threadIdx.x] = 0.f; }
  __syncthreads();

  const int wv   = threadIdx.x >> 5;
  const int lane = threadIdx.x & 31;
  const int hi   = lane >> 4;
  const int l16  = lane & 15;
  const int m0   = blockIdx.y * 64;
  const int n0   = blockIdx.x * 128 + wv * 16;

  if (n0 < N) {
    const f16* xrow = X  + (size_t)(n0 + l16) * CinP + hi * 16;
    const f16* wr0  = Wh + (size_t)(m0 + l16) * CinP + hi * 16;
    const size_t wstep = (size_t)16 * CinP;
    v8f acc[4] = {{}, {}, {}, {}};
    for (int c0 = 0; c0 < CinP; c0 += 32) {
      v16h bf = *(const v16h*)(xrow + c0);       // K = c0+hi*16 .. +15
#pragma unroll
      for (int t = 0; t < 4; ++t) {
        const f16* wr = wr0 + (size_t)t * wstep + c0;
        v8h alo = *(const v8h*)(wr);             // K = c0+hi*16   .. +7
        v8h ahi = *(const v8h*)(wr + 8);         // K = c0+hi*16+8 .. +7
        v16h a  = __builtin_shufflevector(alo, ahi,
                    0, 1, 2, 3, 4, 5, 6, 7, 8, 9, 10, 11, 12, 13, 14, 15);
        acc[t] = __builtin_amdgcn_wmma_f32_16x16x32_f16(false, a, false, bf,
                                                        (short)0, acc[t], false, false);
      }
    }
    // D: lane holds 8 consecutive M = m0+t*16+8*hi+r at row n0+l16
    f16* yrow = Y + (size_t)(n0 + l16) * CoutP + m0 + 8 * hi;
#pragma unroll
    for (int t = 0; t < 4; ++t) {
      v8h d;
#pragma unroll
      for (int r = 0; r < 8; ++r) {
        float v = acc[t][r];
        d[r] = (f16)v;
        atomicAdd(&s_sum[t * 16 + 8 * hi + r], v);
        atomicAdd(&s_sq[t * 16 + 8 * hi + r],  v * v);
      }
      *(v8h*)(yrow + t * 16) = d;
    }
  }
  __syncthreads();
  if (stats && threadIdx.x < 64) {
    int m = m0 + threadIdx.x;
    if (m < Cout) {
      atomicAdd(&stats[m],        s_sum[threadIdx.x]);
      atomicAdd(&stats[Cout + m], s_sq[threadIdx.x]);
    }
  }
}

// ---------------------------------------------------------------------------
// In-place BN (training stats) + ReLU on (NA rows, CP) f16.
// True count = Ncnt rows, C channels; pad rows/channels are forced to zero.
__global__ void k_bn_relu(f16* __restrict__ Y, const float* __restrict__ stats,
                          const float* __restrict__ gamma, const float* __restrict__ beta,
                          int C, int CP, int Ncnt, int NA) {
  size_t i = (size_t)blockIdx.x * blockDim.x + threadIdx.x;
  if (i >= (size_t)NA * CP) return;
  int c = (int)(i % CP);
  int n = (int)(i / CP);
  if (c >= C || n >= Ncnt) { Y[i] = (f16)0.f; return; }
  float inv  = 1.0f / (float)Ncnt;
  float mean = stats[c] * inv;
  float var  = stats[C + c] * inv - mean * mean;
  float x = (float)Y[i];
  float y = gamma[c] * (x - mean) * rsqrtf(var + 1e-5f) + beta[c];
  Y[i] = (f16)fmaxf(y, 0.f);
}

// ---------------------------------------------------------------------------
// Max over K rows: X ((BS*K) rows, CinP) -> Y (BS rows, CoutRow), Y pre-offset.
__global__ void k_maxpool(const f16* __restrict__ X, f16* __restrict__ Y,
                          int C, int CinP, int CoutRow, int BS, int K) {
  int i = blockIdx.x * blockDim.x + threadIdx.x;
  if (i >= BS * C) return;
  int bs = i / C, c = i - bs * C;
  const f16* p = X + (size_t)bs * K * CinP + c;
  float m = -3.0e38f;
  for (int k = 0; k < K; ++k) m = fmaxf(m, (float)p[(size_t)k * CinP]);
  Y[(size_t)bs * CoutRow + c] = (f16)m;
}

// ---------------------------------------------------------------------------
// Final logits: Y (rows=b, CP ch) f16 + bias -> out (B, O) f32.
__global__ void k_finalize(const f16* __restrict__ Y, const float* __restrict__ bias,
                           float* __restrict__ out, int B, int O, int CP) {
  int i = blockIdx.x * blockDim.x + threadIdx.x;
  if (i >= B * O) return;
  int b = i / O, o = i - b * O;
  out[b * O + o] = (float)Y[(size_t)b * CP + o] + bias[o];
}

// ---------------------------------------------------------------------------
extern "C" void kernel_launch(void* const* d_in, const int* in_sizes, int n_in,
                              void* d_out, int out_size, void* d_ws, size_t ws_size,
                              hipStream_t stream) {
  (void)in_sizes; (void)n_in; (void)out_size; (void)ws_size;
  const int B = 8, N0 = 4096, S1 = 512, S2 = 128;

  // ---- unpack inputs (setup_inputs dict order: pc, then params leaves) ----
  int p = 0;
  const float* pc = (const float*)d_in[p++];
  struct Lyr { const float *W, *bi, *g, *be; };
  auto take = [&](bool bn) {
    Lyr L;
    L.W  = (const float*)d_in[p++];
    L.bi = (const float*)d_in[p++];
    L.g  = bn ? (const float*)d_in[p++] : nullptr;
    L.be = bn ? (const float*)d_in[p++] : nullptr;
    return L;
  };
  Lyr sa1[3][3], sa2[3][3], glb[3], cls[3];
  for (int i = 0; i < 3; ++i) for (int j = 0; j < 3; ++j) sa1[i][j] = take(true);
  for (int i = 0; i < 3; ++i) for (int j = 0; j < 3; ++j) sa2[i][j] = take(true);
  for (int i = 0; i < 3; ++i) glb[i] = take(true);
  cls[0] = take(true); cls[1] = take(true); cls[2] = take(false);

  // ---- workspace carve-up ----
  char* wp = (char*)d_ws;
  auto alloc = [&](size_t bytes) -> void* {
    void* r = (void*)wp;
    wp += (bytes + 255) & ~(size_t)255;
    return r;
  };
  float* xyz   = (float*)alloc((size_t)B * N0 * 3 * 4);
  int*   fidx1 = (int*)  alloc((size_t)B * S1 * 4);
  float* xyz1  = (float*)alloc((size_t)B * S1 * 3 * 4);
  int*   fidx2 = (int*)  alloc((size_t)B * S2 * 4);
  float* xyz2  = (float*)alloc((size_t)B * S2 * 3 * 4);
  float* stats = (float*)alloc(2048 * 4);
  f16*   Wh    = (f16*)  alloc((size_t)1024 * 1024 * 2);   // padded f16 weights
  f16*   f1    = (f16*)  alloc((size_t)B * S1 * 320 * 2);  // (4096 rows, 320)
  f16*   f2    = (f16*)  alloc((size_t)B * S2 * 640 * 2);  // (1024 rows, 640)
  f16*   xg    = (f16*)  alloc((size_t)16 * 1024 * 2);     // (16 rows, 1024)
  const size_t BIG = (size_t)B * S1 * 128 * 128 * 2;       // 134 MB
  f16* bufA = (f16*)alloc(BIG);
  f16* bufB = (f16*)alloc(BIG);

  auto r64 = [](int x) { return (x + 63) & ~63; };
  // CinP: row width of X buffer; returns row width of Y buffer (r64(cout)).
  auto run_layer = [&](const Lyr& L, const f16* X, f16* Y, int cout, int cin,
                       int CinP, int Ncnt, int Nrow, bool bn) -> int {
    int CoutP = r64(cout);
    k_cvt_w<<<(CoutP * CinP + 255) / 256, 256, 0, stream>>>(L.W, Wh, cout, cin, CinP, CoutP);
    if (bn) k_zero<<<(2 * cout + 255) / 256, 256, 0, stream>>>(stats, 2 * cout);
    dim3 gg((unsigned)((Nrow + 127) / 128), (unsigned)(CoutP / 64));
    k_gemm16<<<gg, 256, 0, stream>>>(Wh, X, Y, bn ? stats : nullptr, CinP, CoutP, cout, Nrow);
    if (bn) {
      size_t tot = (size_t)Nrow * CoutP;
      k_bn_relu<<<(unsigned)((tot + 255) / 256), 256, 0, stream>>>(Y, stats, L.g, L.be,
                                                                   cout, CoutP, Ncnt, Nrow);
    }
    return CoutP;
  };

  // ---- stage 0: transpose + FPS ----
  k_transpose<<<(B * 3 * N0 + 255) / 256, 256, 0, stream>>>(pc, xyz, B, N0);
  k_fps<<<B, 256, 0, stream>>>(xyz, fidx1, xyz1, N0, S1);

  // ---- SA1 ----
  {
    const int   spec[3][3] = {{32, 32, 64}, {64, 64, 128}, {64, 96, 128}};
    const int   Ks[3]  = {32, 64, 128};
    const float rs[3]  = {0.1f, 0.2f, 0.4f};
    const int   off[3] = {0, 64, 192};
    for (int br = 0; br < 3; ++br) {
      int K = Ks[br], Nt = B * S1 * K;
      k_group_sa1<<<(B * S1) / 8, 256, 0, stream>>>(xyz, xyz1, bufA, B, N0, S1, K,
                                                    rs[br] * rs[br]);
      f16* X = bufA; f16* Y = bufB; int cin = 3, cinP = 32;
      for (int j = 0; j < 3; ++j) {
        cinP = run_layer(sa1[br][j], X, Y, spec[br][j], cin, cinP, Nt, Nt, true);
        cin = spec[br][j];
        f16* t = X; X = Y; Y = t;
      }
      k_maxpool<<<(B * S1 * cin + 255) / 256, 256, 0, stream>>>(
          X, f1 + off[br], cin, cinP, 320, B * S1, K);
    }
  }

  // ---- SA2 ----
  k_fps<<<B, 256, 0, stream>>>(xyz1, fidx2, xyz2, S1, S2);
  {
    const int   spec[3][3] = {{64, 64, 128}, {128, 128, 256}, {128, 128, 256}};
    const int   Ks[3]  = {16, 32, 64};
    const float rs[3]  = {0.2f, 0.4f, 0.8f};
    const int   off[3] = {0, 128, 384};
    for (int br = 0; br < 3; ++br) {
      int K = Ks[br], Nt = B * S2 * K;
      k_group_sa2<<<(B * S2) / 8, 256, 0, stream>>>(xyz1, xyz2, f1, bufA, B, S1, S2, K,
                                                    rs[br] * rs[br]);
      f16* X = bufA; f16* Y = bufB; int cin = 323, cinP = 352;
      for (int j = 0; j < 3; ++j) {
        cinP = run_layer(sa2[br][j], X, Y, spec[br][j], cin, cinP, Nt, Nt, true);
        cin = spec[br][j];
        f16* t = X; X = Y; Y = t;
      }
      k_maxpool<<<(B * S2 * cin + 255) / 256, 256, 0, stream>>>(
          X, f2 + off[br], cin, cinP, 640, B * S2, K);
    }
  }

  // ---- global MLP + global max-pool ----
  int cp;
  cp = run_layer(glb[0], f2,   bufA, 256,  640, 640, B * S2, B * S2, true);
  cp = run_layer(glb[1], bufA, bufB, 512,  256, cp,  B * S2, B * S2, true);
  cp = run_layer(glb[2], bufB, bufA, 1024, 512, cp,  B * S2, B * S2, true);
  k_zero<<<(16 * 1024 * 2 / 4 + 255) / 256, 256, 0, stream>>>((float*)xg, 16 * 1024 / 2);
  k_maxpool<<<(B * 1024 + 255) / 256, 256, 0, stream>>>(bufA, xg, 1024, cp, 1024, B, S2);

  // ---- classifier head (N padded 8 -> 16 rows) ----
  cp = run_layer(cls[0], xg,   bufB, 512, 1024, 1024, B, 16, true);
  cp = run_layer(cls[1], bufB, bufA, 256,  512, cp,   B, 16, true);
  cp = run_layer(cls[2], bufA, bufB,  40,  256, cp,   B, 16, false);
  k_finalize<<<(B * 40 + 255) / 256, 256, 0, stream>>>(bufB, cls[2].bi, (float*)d_out,
                                                       B, 40, 64);
}
